// LatentSelfAttention_57638461112323
// MI455X (gfx1250) — matrix operand order repaired and checked by reading, not compiled, not run on hardware
//
#include <hip/hip_runtime.h>
#include <math.h>

typedef __bf16 bf16_t;
typedef __attribute__((ext_vector_type(16))) __bf16 v16bf;
typedef __attribute__((ext_vector_type(4)))  __bf16 v4bf;
typedef __attribute__((ext_vector_type(8)))  float  v8f;

// GCC-style vector type matching the async-LDS builtin's parameter pointee type
typedef int v4i_vs __attribute__((vector_size(4 * sizeof(int))));
typedef __attribute__((address_space(1))) v4i_vs* gptr_v4i;   // global (device) AS
typedef __attribute__((address_space(3))) v4i_vs* lptr_v4i;   // LDS (shared) AS

#define T_SEQ   2048
#define N_HEAD  16
#define HEAD_D  128
#define Q_COMP  1536
#define KV_COMP 512
#define ROT     64
#define DQK     (KV_COMP + ROT)   // 576

#define BM 128
#define BN 128
#define BK 32
#define LDSP 40   // LDS row pitch in bf16 (32 + 8 pad -> 80B; conflict-free frag reads)

// ---------------------------------------------------------------------------
// CDNA5 async global->LDS copy (ASYNCcnt-tracked, GLOBAL_LOAD_ASYNC_TO_LDS_B128)
// ---------------------------------------------------------------------------
__device__ __forceinline__ void async_g2l_b128(void* lptr, const void* gptr)
{
#if __has_builtin(__builtin_amdgcn_global_load_async_to_lds_b128)
    __builtin_amdgcn_global_load_async_to_lds_b128(
        (gptr_v4i)gptr, (lptr_v4i)lptr, 0, 0);
#else
    unsigned loff =
        (unsigned)(unsigned long long)(__attribute__((address_space(3))) void*)lptr;
    asm volatile("global_load_async_to_lds_b128 %0, %1, off"
                 :: "v"(loff), "v"(gptr) : "memory");
#endif
}

__device__ __forceinline__ void wait_async0()
{
#if __has_builtin(__builtin_amdgcn_s_wait_asynccnt)
    __builtin_amdgcn_s_wait_asynccnt(0);
#else
    asm volatile("s_wait_asynccnt 0" ::: "memory");
#endif
}

// ---------------------------------------------------------------------------
// Tiled WMMA GEMM:  C[M,N] = A[M,K] * B^T (+bias[n]),  B stored [N,K] row-major.
// 128x128 block tile, 8 waves (4 M x 2 N), 32x64 per wave -> 8 WMMA / K-step.
// Double-buffered LDS, async global->LDS staging, one barrier per K-step.
// Batch via blockIdx.z (element strides sA/sB/sC).
// causalSkip: drop tiles fully above the diagonal (scores GEMM).
// kLimitByRow: clamp K loop at m0+BM (P*V GEMM; P zero above diagonal).
// N may be a non-multiple of BN (guarded); M, K must be multiples of BM/BK.
// ---------------------------------------------------------------------------
template <bool OUT_BF16>
__global__ __launch_bounds__(256)
void gemm_wmma(const bf16_t* __restrict__ A, int lda,
               const bf16_t* __restrict__ B, int ldb,
               void* __restrict__ Cv, int ldc,
               int M, int N, int K,
               const float* __restrict__ bias,
               int causalSkip, int kLimitByRow,
               long long sA, long long sB, long long sC)
{
    const int bz = blockIdx.z;
    A += (long long)bz * sA;
    B += (long long)bz * sB;

    const int m0 = blockIdx.y * BM;
    const int n0 = blockIdx.x * BN;
    if (causalSkip && n0 > m0 + BM - 1) return;   // tile fully above diagonal

    __shared__ bf16_t lA[2][BM * LDSP];
    __shared__ bf16_t lB[2][BN * LDSP];

    const int tid  = threadIdx.x;
    const int lane = tid & 31;
    const int wave = tid >> 5;
    const int wm = (wave & 3) * 32;   // wave row offset in tile (4 waves down M)
    const int wn = (wave >> 2) * 64;  // wave col offset in tile (2 waves across N)

    v8f acc[2][4] = {};

    int kEnd = K;
    if (kLimitByRow) { int lim = m0 + BM; kEnd = lim < K ? lim : K; }
    const int nk = kEnd / BK;

    // cooperative staging indices (256 threads; A and B tiles are both 128x32)
    const int arow = tid >> 1;              // 0..127
    const int acol = (tid & 1) * 16;        // 0 / 16
    int bnr = n0 + arow; if (bnr > N - 1) bnr = N - 1;   // clamp partial-N tiles

    const bf16_t* gA = A + (long long)(m0 + arow) * lda + acol;
    const bf16_t* gB = B + (long long)bnr * ldb + acol;

    const int half = lane >> 4;             // 0: lanes 0-15, 1: lanes 16-31
    const int mrow = lane & 15;

    union Frag { v16bf v; uint4 q[2]; };

    // async staging of one K-tile into LDS buffer `buf` (4 x b128 per thread)
    auto stage = [&](int buf, int k) {
        bf16_t* dA = &lA[buf][arow * LDSP + acol];
        bf16_t* dB = &lB[buf][arow * LDSP + acol];
        async_g2l_b128(dA,     gA + k);
        async_g2l_b128(dA + 8, gA + k + 8);
        async_g2l_b128(dB,     gB + k);
        async_g2l_b128(dB + 8, gB + k + 8);
    };

    // ---- prologue: stage tile 0 into buffer 0
    stage(0, 0);
    wait_async0();
    __syncthreads();

    for (int s = 0; s < nk; ++s) {
        const int cur = s & 1;
        const bf16_t* bufA = lA[cur];
        const bf16_t* bufB = lB[cur];

        // kick off async copy of the next tile into the other buffer;
        // it lands in LDS while the WMMAs below execute.
        if (s + 1 < nk) stage(cur ^ 1, (s + 1) * BK);
        if (s + 2 < nk) {   // warm WGP$/L2 two tiles ahead (global_prefetch_b8)
            __builtin_prefetch(gA + (s + 2) * BK, 0, 1);
            __builtin_prefetch(gB + (s + 2) * BK, 0, 1);
        }

        // fragments (16-bit A 16x32 / B 32x16 VGPR striping, ISA 7.12.2)
        Frag fa[2], fb[4];
#pragma unroll
        for (int i = 0; i < 2; ++i) {
            const bf16_t* pa = &bufA[(wm + i * 16 + mrow) * LDSP + half * 8];
            fa[i].q[0] = *(const uint4*)pa;
            fa[i].q[1] = *(const uint4*)(pa + 16);
        }
#pragma unroll
        for (int j = 0; j < 4; ++j) {
            const bf16_t* pb = &bufB[(wn + j * 16 + mrow) * LDSP + half * 8];
            fb[j].q[0] = *(const uint4*)pb;
            fb[j].q[1] = *(const uint4*)(pb + 16);
        }

#pragma unroll
        for (int i = 0; i < 2; ++i)
#pragma unroll
            for (int j = 0; j < 4; ++j)
                acc[i][j] = __builtin_amdgcn_wmma_f32_16x16x32_bf16(
                    false, fa[i].v, false, fb[j].v,
                    (short)0, acc[i][j], false, false);

        wait_async0();       // this wave's async writes to the next buffer done
        __syncthreads();     // all waves done reading cur / writing next
    }

    // ---- epilogue (C layout: vgpr r -> M = r + 8*half, N = lane&15)
    const int crow = half * 8;
    const int ccol = lane & 15;
#pragma unroll
    for (int i = 0; i < 2; ++i) {
#pragma unroll
        for (int j = 0; j < 4; ++j) {
            const int gn = n0 + wn + j * 16 + ccol;
            if (gn >= N) continue;
            const float bval = bias ? bias[gn] : 0.0f;
#pragma unroll
            for (int r = 0; r < 8; ++r) {
                const int gm = m0 + wm + i * 16 + crow + r;
                const float v = acc[i][j][r] + bval;
                if (OUT_BF16)
                    ((bf16_t*)Cv)[(long long)bz * sC + (long long)gm * ldc + gn] = (bf16_t)v;
                else
                    ((float*)Cv)[(long long)bz * sC + (long long)gm * ldc + gn] = v;
            }
        }
    }
}

// ---------------------------------------------------------------------------
// f32 -> bf16 vectorized convert
// ---------------------------------------------------------------------------
__global__ __launch_bounds__(256)
void cvt_f32_bf16(const float* __restrict__ s, bf16_t* __restrict__ d, long long n)
{
    long long i = ((long long)blockIdx.x * 256 + threadIdx.x) * 4;
    if (i + 3 < n) {
        float4 f = *(const float4*)(s + i);
        v4bf o;
        o[0] = (bf16_t)f.x; o[1] = (bf16_t)f.y; o[2] = (bf16_t)f.z; o[3] = (bf16_t)f.w;
        *(v4bf*)(d + i) = o;
    }
}

// f32 [R,C] -> bf16 [C,R] transpose+convert, batched over blockIdx.z
__global__ __launch_bounds__(256)
void transpose_cvt_f32_bf16(const float* __restrict__ src, bf16_t* __restrict__ dst,
                            int R, int C)
{
    __shared__ float tile[32][33];
    const long long bz = blockIdx.z;
    src += bz * (long long)R * C;
    dst += bz * (long long)R * C;
    const int c0 = blockIdx.x * 32, r0 = blockIdx.y * 32;
    const int tx = threadIdx.x & 31, ty = threadIdx.x >> 5;   // 32x8
    for (int i = ty; i < 32; i += 8) {
        const int r = r0 + i, c = c0 + tx;
        tile[i][tx] = (r < R && c < C) ? src[(long long)r * C + c] : 0.0f;
    }
    __syncthreads();
    for (int i = ty; i < 32; i += 8) {
        const int c = c0 + i, r = r0 + tx;
        if (c < C && r < R) dst[(long long)c * R + r] = (bf16_t)tile[tx][i];
    }
}

// bf16 [R,C] -> bf16 [C,R] transpose
__global__ __launch_bounds__(256)
void transpose_bf16(const bf16_t* __restrict__ src, bf16_t* __restrict__ dst,
                    int R, int C)
{
    __shared__ bf16_t tile[32][34];
    const int c0 = blockIdx.x * 32, r0 = blockIdx.y * 32;
    const int tx = threadIdx.x & 31, ty = threadIdx.x >> 5;
    for (int i = ty; i < 32; i += 8) {
        const int r = r0 + i, c = c0 + tx;
        tile[i][tx] = (r < R && c < C) ? src[(long long)r * C + c] : (bf16_t)0.0f;
    }
    __syncthreads();
    for (int i = ty; i < 32; i += 8) {
        const int c = c0 + i, r = r0 + tx;
        if (c < C && r < R) dst[(long long)c * R + r] = tile[tx][i];
    }
}

// ---------------------------------------------------------------------------
// Causal softmax: P[r, 0..r] = softmax(scale*S[r, 0..r]), zeros elsewhere
// ---------------------------------------------------------------------------
__global__ __launch_bounds__(256)
void softmax_causal(const float* __restrict__ S, bf16_t* __restrict__ P,
                    int T, float scale)
{
    const int r = blockIdx.x;
    const float* srow = S + (long long)r * T;
    bf16_t* prow = P + (long long)r * T;
    __shared__ float red[256];
    const int tid = threadIdx.x;

    float mx = -3.0e38f;
    for (int c = tid; c <= r; c += 256) mx = fmaxf(mx, srow[c] * scale);
    red[tid] = mx; __syncthreads();
    for (int s = 128; s > 0; s >>= 1) {
        if (tid < s) red[tid] = fmaxf(red[tid], red[tid + s]);
        __syncthreads();
    }
    mx = red[0]; __syncthreads();

    float sum = 0.0f;
    for (int c = tid; c <= r; c += 256) sum += __expf(srow[c] * scale - mx);
    red[tid] = sum; __syncthreads();
    for (int s = 128; s > 0; s >>= 1) {
        if (tid < s) red[tid] += red[tid + s];
        __syncthreads();
    }
    const float inv = 1.0f / red[0];

    for (int c = tid; c < T; c += 256) {
        float p = (c <= r) ? __expf(srow[c] * scale - mx) * inv : 0.0f;
        prow[c] = (bf16_t)p;
    }
}

// ---------------------------------------------------------------------------
// RoPE for Q_r: qr[T, 16*64] f32 -> Q[h][t][512..575] bf16 (ldq = 576)
// ---------------------------------------------------------------------------
__global__ __launch_bounds__(256)
void rope_q_kernel(const float* __restrict__ qr,
                   const float* __restrict__ cosv, const float* __restrict__ sinv,
                   bf16_t* __restrict__ Q)
{
    int idx = blockIdx.x * 256 + threadIdx.x;     // T * 16 * 32
    if (idx >= T_SEQ * N_HEAD * (ROT / 2)) return;
    const int i = idx & 31;
    const int h = (idx >> 5) & 15;
    const int t = idx >> 9;
    const float x1 = qr[(long long)t * (N_HEAD * ROT) + h * ROT + i];
    const float x2 = qr[(long long)t * (N_HEAD * ROT) + h * ROT + 32 + i];
    const float c = cosv[t * 32 + i], s = sinv[t * 32 + i];
    bf16_t* dst = Q + (long long)h * T_SEQ * DQK + (long long)t * DQK + KV_COMP;
    dst[i]      = (bf16_t)( x1 * c + x2 * s);
    dst[32 + i] = (bf16_t)(-x1 * s + x2 * c);
}

// RoPE for K_r: kr[T,64] f32 -> Kb[t][512..575] bf16 (ldk = 576)
__global__ __launch_bounds__(256)
void rope_k_kernel(const float* __restrict__ kr,
                   const float* __restrict__ cosv, const float* __restrict__ sinv,
                   bf16_t* __restrict__ Kb)
{
    int idx = blockIdx.x * 256 + threadIdx.x;     // T * 32
    if (idx >= T_SEQ * (ROT / 2)) return;
    const int i = idx & 31;
    const int t = idx >> 5;
    const float x1 = kr[(long long)t * ROT + i];
    const float x2 = kr[(long long)t * ROT + 32 + i];
    const float c = cosv[t * 32 + i], s = sinv[t * 32 + i];
    bf16_t* dst = Kb + (long long)t * DQK + KV_COMP;
    dst[i]      = (bf16_t)( x1 * c + x2 * s);
    dst[32 + i] = (bf16_t)(-x1 * s + x2 * c);
}

// ckv bf16 [T,512] -> K buffer cols 0..511 (ldk = 576)
__global__ __launch_bounds__(256)
void copy_ckv_to_k(const bf16_t* __restrict__ ckv, bf16_t* __restrict__ Kb)
{
    int idx = blockIdx.x * 256 + threadIdx.x;     // T * 512
    if (idx >= T_SEQ * KV_COMP) return;
    const int t = idx >> 9;
    const int c = idx & 511;
    Kb[(long long)t * DQK + c] = ckv[idx];
}

// ---------------------------------------------------------------------------
extern "C" void kernel_launch(void* const* d_in, const int* in_sizes, int n_in,
                              void* d_out, int out_size, void* d_ws, size_t ws_size,
                              hipStream_t stream)
{
    (void)in_sizes; (void)n_in; (void)out_size; (void)ws_size;
    const int T = T_SEQ;

    const float* x    = (const float*)d_in[0];
    const float* cosv = (const float*)d_in[1];
    const float* sinv = (const float*)d_in[2];
    const float* Wqd  = (const float*)d_in[3];
    const float* bq   = (const float*)d_in[4];
    const float* Wkvd = (const float*)d_in[5];
    const float* Wqu  = (const float*)d_in[6];
    const float* Wku  = (const float*)d_in[7];
    const float* Wvu  = (const float*)d_in[8];
    const float* Wqr  = (const float*)d_in[9];
    const float* Wkr  = (const float*)d_in[10];
    const float* Wout = (const float*)d_in[11];

    char* p = (char*)d_ws;
    auto take = [&](size_t b) { char* r = p; p += (b + 255) & ~(size_t)255; return r; };

    bf16_t* x_bf    = (bf16_t*)take((size_t)T * 2048 * 2);
    bf16_t* Wqd_bf  = (bf16_t*)take((size_t)Q_COMP * 2048 * 2);
    bf16_t* Wkvd_bf = (bf16_t*)take((size_t)KV_COMP * 2048 * 2);
    bf16_t* Wqu_bf  = (bf16_t*)take((size_t)N_HEAD * HEAD_D * Q_COMP * 2);
    bf16_t* WkuT_bf = (bf16_t*)take((size_t)N_HEAD * KV_COMP * HEAD_D * 2);  // [h][512][128]
    bf16_t* Wvu_bf  = (bf16_t*)take((size_t)N_HEAD * HEAD_D * KV_COMP * 2);
    bf16_t* Wqr_bf  = (bf16_t*)take((size_t)N_HEAD * ROT * Q_COMP * 2);
    bf16_t* Wkr_bf  = (bf16_t*)take((size_t)ROT * 2048 * 2);
    bf16_t* Wout_bf = (bf16_t*)take((size_t)2048 * 2048 * 2);
    bf16_t* cq_bf   = (bf16_t*)take((size_t)T * Q_COMP * 2);
    bf16_t* ckv_bf  = (bf16_t*)take((size_t)T * KV_COMP * 2);
    bf16_t* ckvT_bf = (bf16_t*)take((size_t)KV_COMP * T * 2);               // [512][T]
    float*  krF     = (float*)take((size_t)T * ROT * 4);
    bf16_t* qh_bf   = (bf16_t*)take((size_t)T * 2048 * 2);
    float*  qrF     = (float*)take((size_t)T * N_HEAD * ROT * 4);
    bf16_t* Qbuf    = (bf16_t*)take((size_t)N_HEAD * T * DQK * 2);
    bf16_t* Kbuf    = (bf16_t*)take((size_t)T * DQK * 2);
    float*  scoresF = (float*)take((size_t)T * T * 4);
    bf16_t* probs   = (bf16_t*)take((size_t)T * T * 2);
    bf16_t* attn_bf = (bf16_t*)take((size_t)N_HEAD * T * KV_COMP * 2);      // all heads
    bf16_t* heads_bf= (bf16_t*)take((size_t)T * 2048 * 2);

    auto cvt = [&](const float* s, bf16_t* d, long long n) {
        cvt_f32_bf16<<<(unsigned)((n / 4 + 255) / 256), 256, 0, stream>>>(s, d, n);
    };

    // 1) precision conversion (Wku is transposed per head during conversion)
    cvt(x,    x_bf,    (long long)T * 2048);
    cvt(Wqd,  Wqd_bf,  (long long)Q_COMP * 2048);
    cvt(Wkvd, Wkvd_bf, (long long)KV_COMP * 2048);
    cvt(Wqu,  Wqu_bf,  (long long)N_HEAD * HEAD_D * Q_COMP);
    transpose_cvt_f32_bf16<<<dim3(KV_COMP / 32, HEAD_D / 32, N_HEAD), 256, 0, stream>>>(
        Wku, WkuT_bf, HEAD_D, KV_COMP);
    cvt(Wvu,  Wvu_bf,  (long long)N_HEAD * HEAD_D * KV_COMP);
    cvt(Wqr,  Wqr_bf,  (long long)N_HEAD * ROT * Q_COMP);
    cvt(Wkr,  Wkr_bf,  (long long)ROT * 2048);
    cvt(Wout, Wout_bf, (long long)2048 * 2048);

    // 2) cq = x@Wqd^T + b -> bf16 [T,1536]
    gemm_wmma<true><<<dim3(Q_COMP / BN, T / BM, 1), 256, 0, stream>>>(
        x_bf, 2048, Wqd_bf, 2048, cq_bf, Q_COMP, T, Q_COMP, 2048, bq, 0, 0, 0, 0, 0);

    // 3) ckv = x@Wkvd^T -> bf16 [T,512]; also build ckv^T for the P*V GEMM
    gemm_wmma<true><<<dim3(KV_COMP / BN, T / BM, 1), 256, 0, stream>>>(
        x_bf, 2048, Wkvd_bf, 2048, ckv_bf, KV_COMP, T, KV_COMP, 2048, nullptr, 0, 0, 0, 0, 0);
    transpose_bf16<<<dim3(KV_COMP / 32, T / 32, 1), 256, 0, stream>>>(ckv_bf, ckvT_bf, T, KV_COMP);

    // 4) kr = x@Wkr^T -> f32 [T,64]
    gemm_wmma<false><<<dim3(1, T / BM, 1), 256, 0, stream>>>(
        x_bf, 2048, Wkr_bf, 2048, krF, ROT, T, ROT, 2048, nullptr, 0, 0, 0, 0, 0);

    // 5) qh = cq@Wqu^T -> bf16 [T, 16*128]   ((cq Wqu^T) Wku == cq W_qk, 3x fewer FLOPs)
    gemm_wmma<true><<<dim3(2048 / BN, T / BM, 1), 256, 0, stream>>>(
        cq_bf, Q_COMP, Wqu_bf, Q_COMP, qh_bf, 2048, T, 2048, Q_COMP, nullptr, 0, 0, 0, 0, 0);

    // 6) qr = cq@Wqr^T -> f32 [T, 16*64]
    gemm_wmma<false><<<dim3((N_HEAD * ROT) / BN, T / BM, 1), 256, 0, stream>>>(
        cq_bf, Q_COMP, Wqr_bf, Q_COMP, qrF, N_HEAD * ROT, T, N_HEAD * ROT, Q_COMP,
        nullptr, 0, 0, 0, 0, 0);

    // 7) Q_t[h] = qh[:,h*128:] @ WkuT[h]^T  (batched over heads via blockIdx.z)
    gemm_wmma<true><<<dim3(KV_COMP / BN, T / BM, N_HEAD), 256, 0, stream>>>(
        qh_bf, 2048, WkuT_bf, HEAD_D, Qbuf, DQK, T, KV_COMP, HEAD_D, nullptr, 0, 0,
        (long long)HEAD_D, (long long)KV_COMP * HEAD_D, (long long)T * DQK);

    // 8) RoPE tails + ckv into K head
    rope_q_kernel<<<(T * N_HEAD * (ROT / 2)) / 256, 256, 0, stream>>>(qrF, cosv, sinv, Qbuf);
    rope_k_kernel<<<(T * (ROT / 2)) / 256, 256, 0, stream>>>(krF, cosv, sinv, Kbuf);
    copy_ckv_to_k<<<(T * KV_COMP) / 256, 256, 0, stream>>>(ckv_bf, Kbuf);

    const float scale = 1.0f / sqrtf((float)DQK);

    // 9) per-head attention (scores/probs reused; attn kept per head)
    for (int h = 0; h < N_HEAD; ++h) {
        const bf16_t* Qh = Qbuf + (long long)h * T * DQK;

        // scores = Q_h @ K^T  (upper-triangle tiles skipped)
        gemm_wmma<false><<<dim3(T / BN, T / BM, 1), 256, 0, stream>>>(
            Qh, DQK, Kbuf, DQK, scoresF, T, T, T, DQK, nullptr, 1, 0, 0, 0, 0);

        // probs = causal softmax(scale * scores) -> bf16
        softmax_causal<<<T, 256, 0, stream>>>(scoresF, probs, T, scale);

        // attn_h = probs @ ckv  == probs @ (ckv^T)^T  (K-loop clamped at diagonal)
        gemm_wmma<true><<<dim3(KV_COMP / BN, T / BM, 1), 256, 0, stream>>>(
            probs, T, ckvT_bf, T, attn_bf + (long long)h * T * KV_COMP, KV_COMP,
            T, KV_COMP, T, nullptr, 0, 1, 0, 0, 0);
    }

    // 10) heads[:, h*128:] = attn_h @ Wvu[h]^T  (batched over heads)
    gemm_wmma<true><<<dim3(1, T / BM, N_HEAD), 256, 0, stream>>>(
        attn_bf, KV_COMP, Wvu_bf, KV_COMP, heads_bf, 2048, T, HEAD_D, KV_COMP,
        nullptr, 0, 0,
        (long long)T * KV_COMP, (long long)HEAD_D * KV_COMP, (long long)HEAD_D);

    // 11) out = heads @ Wout^T -> f32 d_out [T,2048]
    gemm_wmma<false><<<dim3(2048 / BN, T / BM, 1), 256, 0, stream>>>(
        heads_bf, 2048, Wout_bf, 2048, (float*)d_out, 2048, T, 2048, 2048,
        nullptr, 0, 0, 0, 0, 0);
}
